// CausalSelfAttention_14113262535146
// MI455X (gfx1250) — compile-verified, matrix operand order checked by reading, not run on hardware
//
#include <hip/hip_runtime.h>

// ---------------------------------------------------------------------------
// Causal self-attention for MI455X (gfx1250, wave32, WMMA, async LDS copies).
// Pipeline:
//   [convert x -> bf16] [transpose+convert W -> bf16 [N][K]]
//   -> GEMM qkv (64x64 wave tile, v_wmma_f32_16x16x32_bf16)
//   -> flash attention (KB=64, online softmax)
//   -> GEMM proj
// All GEMM operands are bf16 row-major-along-K so LDS staging is a straight
// 16B copy (GLOBAL_LOAD_ASYNC_TO_LDS_B128 when available).
// ---------------------------------------------------------------------------

typedef __attribute__((ext_vector_type(16))) __bf16 v16bf;
typedef __attribute__((ext_vector_type(8)))  float  v8f;
typedef int v4i32 __attribute__((vector_size(16)));

#define D_MODEL   1024
#define N_HEAD    16
#define HEAD_DIM  64
#define BATCH     4
#define SEQ       2048
#define MROWS     (BATCH * SEQ)      // 8192 token rows

// ---------------- fp32 -> bf16 (RNE, raw bits) ----------------
__device__ __forceinline__ unsigned short f2bf(float f) {
  unsigned int u = __float_as_uint(f);
  u += 0x7fffu + ((u >> 16) & 1u);
  return (unsigned short)(u >> 16);
}

// ---------------- async global->LDS 16B copy ----------------
#if __has_builtin(__builtin_amdgcn_global_load_async_to_lds_b128)
#define HAVE_ASYNC 1
#else
#define HAVE_ASYNC 0
#endif

__device__ __forceinline__ void cp16(void* lds, const void* g) {
#if HAVE_ASYNC
  __builtin_amdgcn_global_load_async_to_lds_b128(
      (__attribute__((address_space(1))) v4i32*)g,
      (__attribute__((address_space(3))) v4i32*)lds, 0, 0);
#else
  *(uint4*)lds = *(const uint4*)g;
#endif
}

__device__ __forceinline__ void cp_wait() {
#if HAVE_ASYNC
#if __has_builtin(__builtin_amdgcn_s_wait_asynccnt)
  __builtin_amdgcn_s_wait_asynccnt(0);
#else
  asm volatile("s_wait_asynccnt 0" ::: "memory");
#endif
#endif
}

// ---------------- WMMA fragment load from LDS ----------------
// 16x32 bf16 fragment from row-major [16][stride] LDS region.
// A frag: row = M.  B frag: row = N (data pre-transposed so K contiguous).
// CDNA5 16-bit A layout: lanes 0-15: VGPR r<4 -> K=2r,2r+1; r>=4 -> K=16+2(r-4);
// lanes 16-31: +8.
__device__ __forceinline__ v16bf load_frag(const unsigned short* base, int stride) {
  const int lane = threadIdx.x & 31;
  const int hl   = lane >> 4;
  const int row  = lane & 15;
  union { v16bf v; unsigned int u[8]; } f;
#pragma unroll
  for (int r = 0; r < 8; ++r) {
    const int kb = ((r < 4) ? (2 * r) : (16 + 2 * (r - 4))) + 8 * hl;
    f.u[r] = *(const unsigned int*)(base + row * stride + kb);
  }
  return f.v;
}

// ---------------------------------------------------------------------------
// One-shot converters (run once per call; negligible traffic).
// ---------------------------------------------------------------------------
__global__ __launch_bounds__(256)
void cvt_bf16(const float* __restrict__ src, unsigned short* __restrict__ dst, int n) {
  int i = blockIdx.x * 256 + threadIdx.x;
  const int stride = gridDim.x * 256;
  for (; i < n; i += stride) dst[i] = f2bf(src[i]);
}

// W[K][N] fp32  ->  Wt[N][K] bf16   (32x32 LDS tile transpose)
__global__ __launch_bounds__(256)
void transpose_to_bf16(const float* __restrict__ W, unsigned short* __restrict__ Wt,
                       int K, int N) {
  __shared__ float tile[32][33];
  const int k0 = blockIdx.y * 32;
  const int n0 = blockIdx.x * 32;
  const int tx = threadIdx.x & 31;
  const int ty = threadIdx.x >> 5;  // 0..7
#pragma unroll
  for (int i = 0; i < 4; ++i)
    tile[ty + i * 8][tx] = W[(size_t)(k0 + ty + i * 8) * N + n0 + tx];
  __syncthreads();
#pragma unroll
  for (int i = 0; i < 4; ++i)
    Wt[(size_t)(n0 + ty + i * 8) * K + k0 + tx] = f2bf(tile[tx][ty + i * 8]);
}

// ---------------------------------------------------------------------------
// GEMM: Out[M,N] = A[M,K](bf16) @ Bt[N,K](bf16)^T + bias.
// Block tile 128x256x32, 256 threads = 8 waves (2x4), wave tile 64x64
// -> 16 v_wmma per wave per k-step, 128 accumulator VGPRs.
// ---------------------------------------------------------------------------
template <bool OUT_BF16>
__global__ __launch_bounds__(256)
void gemm_bf16_wmma(const unsigned short* __restrict__ A,   // [M][K] bf16
                    const unsigned short* __restrict__ Bt,  // [N][K] bf16
                    const float* __restrict__ bias,
                    void* __restrict__ Out, int M, int N, int K) {
  __shared__ __align__(16) unsigned short As[128 * 32];   // [m][k]
  __shared__ __align__(16) unsigned short Bs[256 * 32];   // [n][k]

  const int tid  = threadIdx.x;
  const int wid  = tid >> 5;
  const int wm   = wid >> 2;          // 0..1 -> 64 rows each
  const int wn   = wid & 3;           // 0..3 -> 64 cols each
  const int m0   = blockIdx.y * 128;
  const int n0   = blockIdx.x * 256;
  const int lane = tid & 31;
  const int hl   = lane >> 4;
  const int c16  = lane & 15;

  v8f acc[4][4];
#pragma unroll
  for (int i = 0; i < 4; ++i)
#pragma unroll
    for (int j = 0; j < 4; ++j)
#pragma unroll
      for (int e = 0; e < 8; ++e) acc[i][j][e] = 0.0f;

  for (int k0 = 0; k0 < K; k0 += 32) {
    __syncthreads();
    // A tile: 128 rows x 64B = 512 x 16B chunks, 2 per thread
#pragma unroll
    for (int i = 0; i < 2; ++i) {
      int c = tid + i * 256;
      int r = c >> 2, off = (c & 3) << 3;
      cp16(As + r * 32 + off, A + (size_t)(m0 + r) * K + k0 + off);
    }
    // B tile: 256 rows x 64B = 1024 x 16B chunks, 4 per thread
#pragma unroll
    for (int i = 0; i < 4; ++i) {
      int c = tid + i * 256;
      int r = c >> 2, off = (c & 3) << 3;
      cp16(Bs + r * 32 + off, Bt + (size_t)(n0 + r) * K + k0 + off);
    }
    cp_wait();
    __syncthreads();

    v16bf a[4], b[4];
#pragma unroll
    for (int i = 0; i < 4; ++i) a[i] = load_frag(As + (wm * 64 + i * 16) * 32, 32);
#pragma unroll
    for (int j = 0; j < 4; ++j) b[j] = load_frag(Bs + (wn * 64 + j * 16) * 32, 32);
#pragma unroll
    for (int i = 0; i < 4; ++i)
#pragma unroll
      for (int j = 0; j < 4; ++j)
        acc[i][j] = __builtin_amdgcn_wmma_f32_16x16x32_bf16(
            false, a[i], false, b[j], (short)0, acc[i][j], false, false);
  }

  // epilogue: C/D layout  M = elem + 8*(lane>=16), N = lane&15
#pragma unroll
  for (int i = 0; i < 4; ++i) {
#pragma unroll
    for (int j = 0; j < 4; ++j) {
      const int rbase = m0 + wm * 64 + i * 16 + 8 * hl;
      const int cbase = n0 + wn * 64 + j * 16 + c16;
      const float bv = bias[cbase];
#pragma unroll
      for (int r = 0; r < 8; ++r) {
        const float v = acc[i][j][r] + bv;
        const size_t off = (size_t)(rbase + r) * N + cbase;
        if (OUT_BF16) ((unsigned short*)Out)[off] = f2bf(v);
        else          ((float*)Out)[off] = v;
      }
    }
  }
}

// ---------------------------------------------------------------------------
// Flash attention: workgroup = (batch, head, 64-query tile), 4 waves, each
// wave owns 16 query rows. Key blocks of 64 -> 16 v_wmma per block iteration.
// qkv bf16 [8192][3072]; y bf16 [8192][1024].
// ---------------------------------------------------------------------------
__global__ __launch_bounds__(128)
void attn_wmma(const unsigned short* __restrict__ qkv, unsigned short* __restrict__ y) {
  __shared__ __align__(16) unsigned short Qs[64 * 64];      // [q][d]
  __shared__ __align__(16) unsigned short Ks[64 * 64];      // [key][d]  (B-frag for QK^T)
  __shared__ __align__(16) unsigned short Vs[64 * 64];      // [d][key]  (B-frag for PV)
  __shared__ __align__(16) unsigned short Ps[4][16 * 64];   // per-wave P (A-frag layout)

  const int tid   = threadIdx.x;
  const int wave  = tid >> 5;
  const int lane  = tid & 31;
  const int hl    = lane >> 4;
  const int c16   = lane & 15;
  const int q0    = blockIdx.x * 64;
  const int head  = blockIdx.y;
  const int batch = blockIdx.z;

  const size_t baseQ = (size_t)batch * SEQ * (3 * D_MODEL) + (size_t)head * HEAD_DIM;
  const size_t baseK = baseQ + D_MODEL;
  const size_t baseV = baseQ + 2 * D_MODEL;

  // ---- stage Q tile 64x64: 512 x 16B chunks / 128 thr = 4 each ----
#pragma unroll
  for (int i = 0; i < 4; ++i) {
    int c = tid + i * 128;
    int qr = c >> 3, off = (c & 7) << 3;
    cp16(Qs + qr * 64 + off, qkv + baseQ + (size_t)(q0 + qr) * (3 * D_MODEL) + off);
  }
  cp_wait();
  __syncthreads();

  const v16bf aQ0 = load_frag(Qs + (wave * 16) * 64 + 0,  64);  // d 0..31
  const v16bf aQ1 = load_frag(Qs + (wave * 16) * 64 + 32, 64);  // d 32..63

  v8f o[4];
#pragma unroll
  for (int n = 0; n < 4; ++n)
#pragma unroll
    for (int e = 0; e < 8; ++e) o[n][e] = 0.0f;
  float mrow[8], lrow[8];
#pragma unroll
  for (int r = 0; r < 8; ++r) { mrow[r] = -1e30f; lrow[r] = 0.0f; }

  const int   qlo   = q0 + wave * 16;
  const int   nkb   = (q0 + 127) >> 6;     // 64-key blocks covering keys 0..q0+63
  const float scale = 0.125f;              // 1/sqrt(64)
  unsigned short* const Pw = &Ps[wave][0];

  for (int kb = 0; kb < nkb; ++kb) {
    const int kstart = kb * 64;
    __syncthreads();                       // guard previous-iter fragment reads
    // ---- stage K tile 64x64 (straight copy) ----
#pragma unroll
    for (int i = 0; i < 4; ++i) {
      int c = tid + i * 128;
      int kr = c >> 3, off = (c & 7) << 3;
      cp16(Ks + kr * 64 + off, qkv + baseK + (size_t)(kstart + kr) * (3 * D_MODEL) + off);
    }
    // ---- stage V tile 64x64, transposed into Vs[d][key] ----
#pragma unroll
    for (int i = 0; i < 16; ++i) {         // 2048 dwords / 128 thr
      int idx = tid + i * 128;
      int kr = idx >> 5;
      int dp = (idx & 31) << 1;
      unsigned int w = *(const unsigned int*)(qkv + baseV + (size_t)(kstart + kr) * (3 * D_MODEL) + dp);
      Vs[(dp + 0) * 64 + kr] = (unsigned short)(w & 0xffffu);
      Vs[(dp + 1) * 64 + kr] = (unsigned short)(w >> 16);
    }
    cp_wait();
    __syncthreads();

    if (kstart <= qlo + 15) {              // causally relevant for this wave
      // ---- S(16x64) = Q @ K^T ----
      v8f s[4];
#pragma unroll
      for (int t = 0; t < 4; ++t) {
#pragma unroll
        for (int e = 0; e < 8; ++e) s[t][e] = 0.0f;
        v16bf bk0 = load_frag(Ks + (t * 16) * 64 + 0,  64);
        v16bf bk1 = load_frag(Ks + (t * 16) * 64 + 32, 64);
        s[t] = __builtin_amdgcn_wmma_f32_16x16x32_bf16(false, aQ0, false, bk0, (short)0, s[t], false, false);
        s[t] = __builtin_amdgcn_wmma_f32_16x16x32_bf16(false, aQ1, false, bk1, (short)0, s[t], false, false);
      }

      // ---- online softmax (row state per C/D element; reduce over 16 lanes) ----
      float alpha[8];
#pragma unroll
      for (int r = 0; r < 8; ++r) {
        const int row = qlo + r + 8 * hl;
        float sv[4];
        bool  ok[4];
#pragma unroll
        for (int t = 0; t < 4; ++t) {
          ok[t] = (kstart + t * 16 + c16) <= row;
          sv[t] = ok[t] ? s[t][r] * scale : -1e30f;
        }
        float vmax = fmaxf(fmaxf(sv[0], sv[1]), fmaxf(sv[2], sv[3]));
#pragma unroll
        for (int msk = 1; msk < 16; msk <<= 1) vmax = fmaxf(vmax, __shfl_xor(vmax, msk, 32));
        const float mnew = fmaxf(mrow[r], vmax);
        float p[4], psum = 0.0f;
#pragma unroll
        for (int t = 0; t < 4; ++t) {
          p[t] = ok[t] ? __expf(sv[t] - mnew) : 0.0f;
          psum += p[t];
        }
#pragma unroll
        for (int msk = 1; msk < 16; msk <<= 1) psum += __shfl_xor(psum, msk, 32);
        const float a = __expf(mrow[r] - mnew);
        lrow[r] = lrow[r] * a + psum;
        mrow[r] = mnew;
        alpha[r] = a;
#pragma unroll
        for (int t = 0; t < 4; ++t)
          Pw[(r + 8 * hl) * 64 + t * 16 + c16] = f2bf(p[t]);
      }
      // rescale running output
#pragma unroll
      for (int n = 0; n < 4; ++n)
#pragma unroll
        for (int r = 0; r < 8; ++r) o[n][r] *= alpha[r];

      // same-wave LDS RAW on Ps: drain DS stores before fragment reload
      asm volatile("s_wait_dscnt 0" ::: "memory");

      // ---- O(16x64) += P(16x64) @ V(64x64) ----
      const v16bf aP0 = load_frag(Pw + 0,  64);
      const v16bf aP1 = load_frag(Pw + 32, 64);
#pragma unroll
      for (int n = 0; n < 4; ++n) {
        v16bf bv0 = load_frag(Vs + (n * 16) * 64 + 0,  64);
        v16bf bv1 = load_frag(Vs + (n * 16) * 64 + 32, 64);
        o[n] = __builtin_amdgcn_wmma_f32_16x16x32_bf16(false, aP0, false, bv0, (short)0, o[n], false, false);
        o[n] = __builtin_amdgcn_wmma_f32_16x16x32_bf16(false, aP1, false, bv1, (short)0, o[n], false, false);
      }
    }
  }

  // ---- epilogue: y[b, q, head*64 + d] = O / l (bf16) ----
#pragma unroll
  for (int r = 0; r < 8; ++r) {
    const float inv = (lrow[r] > 0.0f) ? (1.0f / lrow[r]) : 0.0f;
    const int row = qlo + r + 8 * hl;
    const size_t rb = ((size_t)batch * SEQ + row) * D_MODEL + (size_t)head * HEAD_DIM;
#pragma unroll
    for (int n = 0; n < 4; ++n) {
      y[rb + n * 16 + c16] = f2bf(o[n][r] * inv);
    }
  }
}

// ---------------------------------------------------------------------------
// Host-side launcher
// ---------------------------------------------------------------------------
extern "C" void kernel_launch(void* const* d_in, const int* in_sizes, int n_in,
                              void* d_out, int out_size, void* d_ws, size_t ws_size,
                              hipStream_t stream) {
  (void)in_sizes; (void)n_in; (void)out_size; (void)ws_size;
  const float* x      = (const float*)d_in[0];   // [4,2048,1024]
  const float* W_attn = (const float*)d_in[1];   // [1024,3072]
  const float* b_attn = (const float*)d_in[2];   // [3072]
  const float* W_proj = (const float*)d_in[3];   // [1024,1024]
  const float* b_proj = (const float*)d_in[4];   // [1024]
  float* out = (float*)d_out;                    // [4,2048,1024]

  // workspace layout (bf16 elements); xb's region is reused for y after GEMM1
  unsigned short* qkv = (unsigned short*)d_ws;                     // 8192 x 3072  (48 MB)
  unsigned short* Wta = qkv + (size_t)MROWS * (3 * D_MODEL);       // 3072 x 1024  ( 6 MB)
  unsigned short* Wtp = Wta + (size_t)(3 * D_MODEL) * D_MODEL;     // 1024 x 1024  ( 2 MB)
  unsigned short* xb  = Wtp + (size_t)D_MODEL * D_MODEL;           // 8192 x 1024  (16 MB)
  unsigned short* yws = xb;                                        // aliases xb (x dead after GEMM1)

  // 0) one-shot conversions
  cvt_bf16<<<2048, 256, 0, stream>>>(x, xb, MROWS * D_MODEL);
  transpose_to_bf16<<<dim3((3 * D_MODEL) / 32, D_MODEL / 32), 256, 0, stream>>>(
      W_attn, Wta, D_MODEL, 3 * D_MODEL);
  transpose_to_bf16<<<dim3(D_MODEL / 32, D_MODEL / 32), 256, 0, stream>>>(
      W_proj, Wtp, D_MODEL, D_MODEL);

  // 1) qkv = x @ W_attn + b_attn   (bf16 -> bf16)
  gemm_bf16_wmma<true><<<dim3((3 * D_MODEL) / 256, MROWS / 128), 256, 0, stream>>>(
      xb, Wta, b_attn, qkv, MROWS, 3 * D_MODEL, D_MODEL);

  // 2) flash attention -> y (bf16)
  attn_wmma<<<dim3(SEQ / 64, N_HEAD, BATCH), 128, 0, stream>>>(qkv, yws);

  // 3) out = y @ W_proj + b_proj   (bf16 -> fp32)
  gemm_bf16_wmma<false><<<dim3(D_MODEL / 256, MROWS / 128), 256, 0, stream>>>(
      yws, Wtp, b_proj, out, MROWS, D_MODEL, D_MODEL);
}
